// SpatioTemporalHyperGNN_50096498540670
// MI455X (gfx1250) — matrix operand order repaired and checked by reading, not compiled, not run on hardware
//
#include <hip/hip_runtime.h>
#include <hip/hip_bf16.h>
#include <math.h>

// ---------------------------------------------------------------------------
// Problem constants (from the reference)
// ---------------------------------------------------------------------------
static constexpr int kT  = 4;
static constexpr int kN  = 2048;   // nodes
static constexpr int kE  = 512;    // hyperedges
static constexpr int kF  = 256;    // input features
static constexpr int kH  = 256;    // hidden
static constexpr int kH2 = 512;    // 2*hidden
static constexpr int kX  = 8;      // experts

// ---------------------------------------------------------------------------
// WMMA vector types (CDNA5 wave32 WMMA, 16x16x32 bf16 -> f32)
// ---------------------------------------------------------------------------
typedef __attribute__((ext_vector_type(16))) __bf16 v16bf;
typedef __attribute__((ext_vector_type(8)))  __bf16 v8bf;
typedef __attribute__((ext_vector_type(4)))  __bf16 v4bf;
typedef __attribute__((ext_vector_type(8)))  float  v8f;

__device__ __forceinline__ float gelu_erf(float x) {
    return 0.5f * x * (1.0f + erff(x * 0.70710678118654752440f));
}

// ---------------------------------------------------------------------------
// bf16 WMMA GEMM:  C[M,Nc] = epi(A[M,K] @ B[K,Nc] [+ bias])
// A, B bf16 row-major in global (pre-converted; tiles stage to LDS as raw
// b128 copies -- no converts in the hot loop).
// EPI: 0 = store bf16, 1 = gelu -> bf16, 2 = Cf += wrow[row]*val (fp32 acc)
// Tile 128x128x32; 256 threads = 8 waves; wave = 2x4 WMMA tiles.
// Software-pipelined: next tile staged in registers while WMMA consumes LDS.
// ---------------------------------------------------------------------------
#define BM 128
#define BN 128
#define BK 32

template <int EPI, bool HAS_BIAS>
__global__ __launch_bounds__(256)
void gemm_bf16_wmma(const __bf16* __restrict__ A,
                    const __bf16* __restrict__ B,
                    const float* __restrict__ bias,
                    __bf16* __restrict__ Cb,      // EPI 0/1 output
                    float* __restrict__ Cf,       // EPI 2 accumulator
                    const float* __restrict__ wrow, int wstride,
                    int M, int Nc, int K) {
    __shared__ alignas(32) __bf16 Asm[BM][BK];   // row-major, 8 KB
    __shared__ alignas(32) __bf16 Bsm[BN][BK];   // column-major (col,k), 8 KB

    const int tid   = threadIdx.x;
    const int wave  = tid >> 5;
    const int lane  = tid & 31;
    const int wm    = wave >> 1;        // 0..3 (wave row)
    const int wn    = wave & 1;         // 0..1 (wave col)
    const int lhalf = lane >> 4;        // 0/1 (K-half select / M+8 select)
    const int l16   = lane & 15;

    const int bm = blockIdx.y * BM;
    const int bn = blockIdx.x * BN;

    // global->LDS staging indices
    const int ar = tid >> 1;            // A: row 0..127, 2 threads/row
    const int ac = (tid & 1) * 16;      //    16 bf16 each
    const int bk = tid >> 3;            // B: k-row 0..31, 8 threads/row
    const int bc = (tid & 7) * 16;      //    16 cols each

    // running global pointers (strength-reduced across the K loop)
    const __bf16* Ap = A + (size_t)(bm + ar) * K + ac;
    const __bf16* Bp = B + (size_t)bk * Nc + bn + bc;
    const size_t bstep = (size_t)BK * Nc;

    v8f acc[2][4];
#pragma unroll
    for (int mi = 0; mi < 2; ++mi)
#pragma unroll
        for (int ni = 0; ni < 4; ++ni)
            acc[mi][ni] = (v8f){0.f, 0.f, 0.f, 0.f, 0.f, 0.f, 0.f, 0.f};

    uint4 areg0, areg1;
    union BStage { uint4 u[2]; __bf16 e[16]; } breg;

    auto load_tile = [&]() {
        areg0 = *reinterpret_cast<const uint4*>(Ap);
        areg1 = *reinterpret_cast<const uint4*>(Ap + 8);
        breg.u[0] = *reinterpret_cast<const uint4*>(Bp);
        breg.u[1] = *reinterpret_cast<const uint4*>(Bp + 8);
        Ap += BK;
        Bp += bstep;
    };

    auto stage_lds = [&]() {
        *reinterpret_cast<uint4*>(&Asm[ar][ac])     = areg0;
        *reinterpret_cast<uint4*>(&Asm[ar][ac + 8]) = areg1;
#pragma unroll
        for (int q = 0; q < 16; ++q)
            Bsm[bc + q][bk] = breg.e[q];   // transpose scatter
    };

    auto compute_tile = [&]() {
        // A 16x32: lanes 0-15 row M, K 0..15; lanes 16-31 K 16..31
        v16bf afrag[2];
#pragma unroll
        for (int mi = 0; mi < 2; ++mi) {
            int row = wm * 32 + mi * 16 + l16;
            v8bf lo = *reinterpret_cast<const v8bf*>(&Asm[row][lhalf * 16]);
            v8bf hi = *reinterpret_cast<const v8bf*>(&Asm[row][lhalf * 16 + 8]);
            afrag[mi] = __builtin_shufflevector(lo, hi, 0, 1, 2, 3, 4, 5, 6, 7,
                                                8, 9, 10, 11, 12, 13, 14, 15);
        }
        // B 32x16: lanes 0-15 col N, K 0..15; lanes 16-31 K 16..31
        v16bf bfrag[4];
#pragma unroll
        for (int ni = 0; ni < 4; ++ni) {
            int col = wn * 64 + ni * 16 + l16;
            bfrag[ni] = *reinterpret_cast<const v16bf*>(&Bsm[col][lhalf * 16]);
        }
#pragma unroll
        for (int mi = 0; mi < 2; ++mi)
#pragma unroll
            for (int ni = 0; ni < 4; ++ni)
                acc[mi][ni] = __builtin_amdgcn_wmma_f32_16x16x32_bf16(
                    false, afrag[mi], false, bfrag[ni],
                    (short)0, acc[mi][ni], false, false);
    };

    // ---- software pipeline: regs(k+1) in flight while LDS(k) is consumed ----
    load_tile();
    stage_lds();
    __syncthreads();
    for (int k0 = BK; k0 < K; k0 += BK) {
        if (k0 + BK < K) {  // pull tile-after-next toward L2/L0
            __builtin_prefetch(Ap + BK, 0, 1);
            __builtin_prefetch(Bp + bstep, 0, 1);
        }
        load_tile();        // global loads for tile k0 issue here...
        compute_tile();     // ...and overlap with WMMA on tile k0-BK
        __syncthreads();
        stage_lds();
        __syncthreads();
    }
    compute_tile();

    // ---- epilogue (C/D layout: VGPR v -> M = v + 8*lhalf, N = l16) ----
#pragma unroll
    for (int mi = 0; mi < 2; ++mi) {
        const int row0 = bm + wm * 32 + mi * 16 + lhalf * 8;
#pragma unroll
        for (int ni = 0; ni < 4; ++ni) {
            const int col = bn + wn * 64 + ni * 16 + l16;
            const float bv = HAS_BIAS ? bias[col] : 0.0f;
            if (EPI == 2) {
                float* cp = Cf + (size_t)row0 * Nc + col;
                const float* wp = wrow + (size_t)row0 * wstride;
#pragma unroll
                for (int v = 0; v < 8; ++v) {
                    float val = acc[mi][ni][v] + bv;
                    *cp += wp[(size_t)v * wstride] * val;
                    cp += Nc;
                }
            } else {
                __bf16* cp = Cb + (size_t)row0 * Nc + col;
#pragma unroll
                for (int v = 0; v < 8; ++v) {
                    float val = acc[mi][ni][v] + bv;
                    if (EPI == 1) val = gelu_erf(val);
                    *cp = (__bf16)val;
                    cp += Nc;
                }
            }
        }
    }
}

// ---------------------------------------------------------------------------
// Small helper kernels
// ---------------------------------------------------------------------------
// fp32 -> bf16 (vectorized x4; all counts are multiples of 4)
__global__ void cvt_bf16_kernel(const float* __restrict__ src,
                                __bf16* __restrict__ dst, int count) {
    int i = (blockIdx.x * 256 + threadIdx.x) * 4;
    if (i + 3 < count) {
        float4 f = *reinterpret_cast<const float4*>(src + i);
        v4bf p = {(__bf16)f.x, (__bf16)f.y, (__bf16)f.z, (__bf16)f.w};
        *reinterpret_cast<v4bf*>(dst + i) = p;
    }
}

__global__ void dv_kernel(const float* __restrict__ H, float* __restrict__ dv) {
    __shared__ float red[256];
    int n = blockIdx.x, tid = threadIdx.x;
    float s = 0.f;
    for (int e = tid; e < kE; e += 256) s += H[(size_t)n * kE + e];
    red[tid] = s; __syncthreads();
    for (int o = 128; o > 0; o >>= 1) {
        if (tid < o) red[tid] += red[tid + o];
        __syncthreads();
    }
    if (tid == 0) dv[n] = rsqrtf(fmaxf(red[0], 1.0f));
}

__global__ void de_kernel(const float* __restrict__ H, float* __restrict__ de) {
    __shared__ float red[256];
    int e = blockIdx.x, tid = threadIdx.x;
    float s = 0.f;
    for (int n = tid; n < kN; n += 256) s += H[(size_t)n * kE + e];
    red[tid] = s; __syncthreads();
    for (int o = 128; o > 0; o >>= 1) {
        if (tid < o) red[tid] += red[tid + o];
        __syncthreads();
    }
    if (tid == 0) de[e] = 1.0f / fmaxf(red[0], 1.0f);
}

// A[n,e] = dv[n] * H[n,e] * Wdiag[e] * de[e]   (bf16 out)
__global__ void buildA_kernel(const float* __restrict__ H, const float* __restrict__ Wt,
                              const float* __restrict__ dv, const float* __restrict__ de,
                              __bf16* __restrict__ As) {
    int idx = blockIdx.x * 256 + threadIdx.x;   // < kN*kE
    int n = idx / kE, e = idx % kE;
    As[idx] = (__bf16)(dv[n] * H[idx] * Wt[(size_t)e * kE + e] * de[e]);
}

// B[e,n] = H[n,e] * dv[n]   (bf16 out)
__global__ void buildB_kernel(const float* __restrict__ H, const float* __restrict__ dv,
                              __bf16* __restrict__ Bs) {
    int idx = blockIdx.x * 256 + threadIdx.x;   // < kE*kN
    int e = idx / kN, n = idx % kN;
    Bs[idx] = (__bf16)(H[(size_t)n * kE + e] * dv[n]);
}

// softmax(h2@wr+br) -> top-2 renormalized combine weights (jax tie semantics)
__global__ void router_kernel(const __bf16* __restrict__ h2, const float* __restrict__ wr,
                              const float* __restrict__ br, float* __restrict__ wcomb) {
    int n = blockIdx.x * blockDim.x + threadIdx.x;
    if (n >= kN) return;
    float l[kX];
#pragma unroll
    for (int k = 0; k < kX; ++k) l[k] = br[k];
    const __bf16* hrow = h2 + (size_t)n * kH;
    for (int j = 0; j < kH; ++j) {
        float hv = (float)hrow[j];
        const float* wj = wr + (size_t)j * kX;
#pragma unroll
        for (int k = 0; k < kX; ++k) l[k] += hv * wj[k];
    }
    float m = l[0];
#pragma unroll
    for (int k = 1; k < kX; ++k) m = fmaxf(m, l[k]);
    float p[kX], s = 0.f;
#pragma unroll
    for (int k = 0; k < kX; ++k) { p[k] = expf(l[k] - m); s += p[k]; }
    float inv = 1.0f / s;
#pragma unroll
    for (int k = 0; k < kX; ++k) p[k] *= inv;
    int i1 = 0;
#pragma unroll
    for (int k = 1; k < kX; ++k) if (p[k] > p[i1]) i1 = k;
    int i2 = (i1 == 0) ? 1 : 0;
#pragma unroll
    for (int k = 0; k < kX; ++k) if (k != i1 && p[k] > p[i2]) i2 = k;
    float norm = 1.0f / (p[i1] + p[i2]);
#pragma unroll
    for (int k = 0; k < kX; ++k)
        wcomb[(size_t)n * kX + k] =
            (k == i1) ? p[i1] * norm : ((k == i2) ? p[i2] * norm : 0.0f);
}

__global__ void risk_kernel(const float* __restrict__ ssum, const float* __restrict__ w_risk,
                            const float* __restrict__ b_risk, float* __restrict__ out) {
    int n = blockIdx.x * blockDim.x + threadIdx.x;
    if (n >= kN) return;
    float s = b_risk[0];
    const float* row = ssum + (size_t)n * kH;
    for (int j = 0; j < kH; ++j) s += (row[j] * (1.0f / kT)) * w_risk[j];
    out[n] = 1.0f / (1.0f + expf(-s));
}

__global__ void zero_kernel(float* __restrict__ p, int count) {
    int i = blockIdx.x * blockDim.x + threadIdx.x;
    if (i < count) p[i] = 0.0f;
}

// ---------------------------------------------------------------------------
// Launcher
// ---------------------------------------------------------------------------
extern "C" void kernel_launch(void* const* d_in, const int* in_sizes, int n_in,
                              void* d_out, int out_size, void* d_ws, size_t ws_size,
                              hipStream_t stream) {
    const float* x_seq  = (const float*)d_in[0];   // [1,T,N,F]
    const float* H_seq  = (const float*)d_in[1];   // [T,N,E]
    const float* W_seq  = (const float*)d_in[2];   // [T,E,E]
    const float* w1     = (const float*)d_in[3];   // [F,H]
    const float* b1     = (const float*)d_in[4];   // [H]
    const float* w2     = (const float*)d_in[5];   // [H,H]
    const float* b2     = (const float*)d_in[6];   // [H]
    const float* wr     = (const float*)d_in[7];   // [H,X]
    const float* br     = (const float*)d_in[8];   // [X]
    const float* We1    = (const float*)d_in[9];   // [X,H,2H]
    const float* be1    = (const float*)d_in[10];  // [X,2H]
    const float* We2    = (const float*)d_in[11];  // [X,2H,H]
    const float* be2    = (const float*)d_in[12];  // [X,H]
    const float* w_risk = (const float*)d_in[13];  // [H,1]
    const float* b_risk = (const float*)d_in[14];  // [1]
    float* out = (float*)d_out;                    // [N,1]

    // workspace carve-up: fp32 first, then bf16
    char* base = (char*)d_ws;
    auto alloc_f = [&](size_t n) { float* p = (float*)base;  base += n * sizeof(float);  return p; };
    auto alloc_b = [&](size_t n) { __bf16* p = (__bf16*)base; base += n * sizeof(__bf16); return p; };

    float*  dv    = alloc_f(kN);
    float*  de    = alloc_f(kE);
    float*  wcomb = alloc_f((size_t)kN * kX);
    float*  ssum  = alloc_f((size_t)kN * kH);
    __bf16* A_s   = alloc_b((size_t)kN * kE);
    __bf16* B_s   = alloc_b((size_t)kE * kN);
    __bf16* xtb   = alloc_b((size_t)kN * kF);
    __bf16* XWb   = alloc_b((size_t)kN * kH);    // also reused for h1@w2+b2
    __bf16* Pb    = alloc_b((size_t)kE * kH);
    __bf16* h1b   = alloc_b((size_t)kN * kH);
    __bf16* h2b   = alloc_b((size_t)kN * kH);
    __bf16* T1b   = alloc_b((size_t)kN * kH2);
    __bf16* w1b   = alloc_b((size_t)kF * kH);
    __bf16* w2b   = alloc_b((size_t)kH * kH);
    __bf16* We1b  = alloc_b((size_t)kX * kH * kH2);
    __bf16* We2b  = alloc_b((size_t)kX * kH2 * kH);

    const dim3 blk(256);
    auto ggrid = [](int M, int Nc) { return dim3(Nc / BN, M / BM); };
    auto cvt = [&](const float* s, __bf16* d, int count) {
        cvt_bf16_kernel<<<(count / 4 + 255) / 256, blk, 0, stream>>>(s, d, count);
    };

    // one-time (per launch) weight conversion to bf16
    cvt(w1, w1b, kF * kH);
    cvt(w2, w2b, kH * kH);
    cvt(We1, We1b, kX * kH * kH2);
    cvt(We2, We2b, kX * kH2 * kH);

    // zero the temporal/MoE accumulator
    zero_kernel<<<(kN * kH + 255) / 256, blk, 0, stream>>>(ssum, kN * kH);

    for (int t = 0; t < kT; ++t) {
        const float* Ht = H_seq + (size_t)t * kN * kE;
        const float* Wt = W_seq + (size_t)t * kE * kE;
        const float* xt = x_seq + (size_t)t * kN * kF;

        cvt(xt, xtb, kN * kF);
        dv_kernel<<<kN, blk, 0, stream>>>(Ht, dv);
        de_kernel<<<kE, blk, 0, stream>>>(Ht, de);
        buildA_kernel<<<(kN * kE) / 256, blk, 0, stream>>>(Ht, Wt, dv, de, A_s);
        buildB_kernel<<<(kE * kN) / 256, blk, 0, stream>>>(Ht, dv, B_s);

        // conv1: h1 = gelu(A @ (B @ (x@w1 + b1)))
        gemm_bf16_wmma<0, true><<<ggrid(kN, kH), blk, 0, stream>>>(
            xtb, w1b, b1, XWb, nullptr, nullptr, 0, kN, kH, kF);
        gemm_bf16_wmma<0, false><<<ggrid(kE, kH), blk, 0, stream>>>(
            B_s, XWb, nullptr, Pb, nullptr, nullptr, 0, kE, kH, kN);
        gemm_bf16_wmma<1, false><<<ggrid(kN, kH), blk, 0, stream>>>(
            A_s, Pb, nullptr, h1b, nullptr, nullptr, 0, kN, kH, kE);

        // conv2: h2 = gelu(A @ (B @ (h1@w2 + b2)))
        gemm_bf16_wmma<0, true><<<ggrid(kN, kH), blk, 0, stream>>>(
            h1b, w2b, b2, XWb, nullptr, nullptr, 0, kN, kH, kH);
        gemm_bf16_wmma<0, false><<<ggrid(kE, kH), blk, 0, stream>>>(
            B_s, XWb, nullptr, Pb, nullptr, nullptr, 0, kE, kH, kN);
        gemm_bf16_wmma<1, false><<<ggrid(kN, kH), blk, 0, stream>>>(
            A_s, Pb, nullptr, h2b, nullptr, nullptr, 0, kN, kH, kE);

        // MoE routing (top-2, renormalized)
        router_kernel<<<(kN + 255) / 256, blk, 0, stream>>>(h2b, wr, br, wcomb);

        // dense expert loop; combine fused into second GEMM epilogue,
        // accumulated straight into the temporal sum (deterministic order).
        for (int e = 0; e < kX; ++e) {
            gemm_bf16_wmma<1, true><<<ggrid(kN, kH2), blk, 0, stream>>>(
                h2b, We1b + (size_t)e * kH * kH2, be1 + (size_t)e * kH2,
                T1b, nullptr, nullptr, 0, kN, kH2, kH);
            gemm_bf16_wmma<2, true><<<ggrid(kN, kH), blk, 0, stream>>>(
                T1b, We2b + (size_t)e * kH2 * kH, be2 + (size_t)e * kH,
                nullptr, ssum, wcomb + e, kX, kN, kH, kH2);
        }
    }

    // risk = sigmoid(mean_t(states) @ w_risk + b_risk)
    risk_kernel<<<(kN + 255) / 256, blk, 0, stream>>>(ssum, w_risk, b_risk, out);
}
